// DistogramDecoder_4063039062799
// MI455X (gfx1250) — compile-verified, matrix operand order checked by reading, not run on hardware
//
#include <hip/hip_runtime.h>
#include <hip/hip_bf16.h>

// ---------------------------------------------------------------------------
// DistogramDecoder for MI455X (gfx1250, wave32, WMMA 16x16x32 f16)
//   N=8192 nodes, K=32 neighbours, P=128, BINS=64
//   All GEMMs via v_wmma_f32_16x16x32_f16, f32 accumulate.
//   B-matrices pre-swizzled to fragment-major order -> b128 loads only.
// ---------------------------------------------------------------------------

typedef __attribute__((ext_vector_type(16))) _Float16 v16h;
typedef __attribute__((ext_vector_type(8)))  float    v8f;

#define N_NODES  8192
#define KNBR     32
#define DLOC     256
#define PP       128
#define BINS     64
#define RELCLIP  32
#define RELDIM   66

// workspace layout (bytes)  -- all B-weights stored fragment-major
#define WS_WFEAT 0u         // 64x128 f16     (16384 B)
#define WS_W1    16384u     // 128x256 f16    (65536 B)
#define WS_W2    81920u     // 256x64  f16    (32768 B)
#define WS_WLOC  114688u    // 256x256 f16    (131072 B)  [W_loc_i | W_loc_j]
#define WS_R     245760u    // 8192x9 f32     (294912 B)
#define WS_LOCIJ 540672u    // 8192x256 f32   (8388608 B)
#define WS_TOTAL 8929280u

// fused-kernel LDS layout (bytes)
#define SM_WFEAT 0u
#define SM_W1    16384u
#define SM_W2    81920u
#define SM_FEATA 114688u    // 32x64 f16
#define SM_PAIRF 118784u    // 32x128 f32
#define SM_PAIRH 135168u    // 32x128 f16
#define SM_HID   143360u    // 32x256 f16
#define SM_LOGIT 159744u    // 32x64 f32
#define SM_J     167936u    // 32 int
#define SM_CODE  168064u    // 32 int
#define SM_PM    168192u    // 32 f32
#define SM_BYTES 168448u

// bytes of packed f16 weights staged to LDS each block (Wfeat+W1+W2)
#define STAGE_BYTES 114688
#define STAGE_VEC16 (STAGE_BYTES / 16)

// ---------------------------------------------------------------------------
// WMMA fragment helpers (ISA 7.12.2 layouts, wave32)
//   A 16x32 f16:  lane l -> row M = l&15; halves j<8 -> K = kb+j,
//                 j>=8 -> K = kb+j+8, with kb = k0 + ((l>=16)?8:0)
//   B fragment-major storage: block blk holds 32 lanes x 16 halves,
//       lane l's v16h at base + blk*512 + l*16  (32 contiguous bytes)
//   C/D 16x16 f32: reg r -> M = m0 + r + ((l>=16)?8:0), N = n0+(l&15)
// ---------------------------------------------------------------------------
__device__ __forceinline__ v16h frag_a(const _Float16* base, int stride,
                                       int m0, int k0, int lane) {
  v16h a;
  const _Float16* p = base + (m0 + (lane & 15)) * stride + k0 + ((lane >> 4) << 3);
#pragma unroll
  for (int j = 0; j < 8; ++j) a[j] = p[j];
#pragma unroll
  for (int j = 8; j < 16; ++j) a[j] = p[j + 8];
  return a;
}

__device__ __forceinline__ v16h frag_b_swz(const _Float16* base, int blk, int lane) {
  const _Float16* p = base + (size_t)blk * 512 + lane * 16;
  v16h b;
#pragma unroll
  for (int j = 0; j < 16; ++j) b[j] = p[j];   // vectorizes to 2x b128
  return b;
}

__device__ __forceinline__ v8f wmma16(v16h a, v16h b, v8f c) {
  return __builtin_amdgcn_wmma_f32_16x16x32_f16(false, a, false, b,
                                                (short)0, c, false, false);
}

__device__ __forceinline__ v8f zero8() {
  v8f z = {0.f, 0.f, 0.f, 0.f, 0.f, 0.f, 0.f, 0.f};
  return z;
}

// decode fragment-major dest index d (elements) -> (k, n) of a KxN matrix
//   blk = nt * (K/32) + ks
__device__ __forceinline__ void swz_kn(int d, int K, int N, int* k, int* n) {
  int f = d >> 9;            // /512 elements per fragment block
  int r = d & 511;
  int lane = r >> 4, j = r & 15;
  int numKS = K >> 5;
  int nt = f / numKS, ks = f - nt * numKS;
  *k = ks * 32 + ((lane >> 4) << 3) + j + ((j >= 8) ? 8 : 0);
  *n = nt * 16 + (lane & 15);
}

// ---------------------------------------------------------------------------
// Kernel 1: pack / convert weights to f16 fragment-major in workspace
// ---------------------------------------------------------------------------
__global__ void prep_weights(const float* Wdist, const float* Wdir,
                             const float* Wrot, const float* Wpvec,
                             const float* Wloci, const float* Wlocj,
                             const float* W1, const float* W2, char* ws) {
  int tid = blockIdx.x * blockDim.x + threadIdx.x;
  int nthr = gridDim.x * blockDim.x;
  int k, n;

  // Wfeat: composite 64x128; rows 0-15 dist, 16-18 dir, 19-27 rot, 28-42 pvec
  _Float16* wfeat = (_Float16*)(ws + WS_WFEAT);
  for (int d = tid; d < 64 * PP; d += nthr) {
    swz_kn(d, 64, PP, &k, &n);
    float v = 0.f;
    if (k < 16)      v = Wdist[k * PP + n];
    else if (k < 19) v = Wdir[(k - 16) * PP + n];
    else if (k < 28) v = Wrot[(k - 19) * PP + n];
    else if (k < 43) v = Wpvec[(k - 28) * PP + n];
    wfeat[d] = (_Float16)v;
  }
  _Float16* w1h = (_Float16*)(ws + WS_W1);          // 128x256
  for (int d = tid; d < 128 * 256; d += nthr) {
    swz_kn(d, 128, 256, &k, &n);
    w1h[d] = (_Float16)W1[k * 256 + n];
  }
  _Float16* w2h = (_Float16*)(ws + WS_W2);          // 256x64
  for (int d = tid; d < 256 * 64; d += nthr) {
    swz_kn(d, 256, 64, &k, &n);
    w2h[d] = (_Float16)W2[k * 64 + n];
  }
  _Float16* wl = (_Float16*)(ws + WS_WLOC);         // 256x256 [Wloci|Wlocj]
  for (int d = tid; d < 256 * 256; d += nthr) {
    swz_kn(d, 256, 256, &k, &n);
    wl[d] = (_Float16)((n < 128) ? Wloci[k * 128 + n] : Wlocj[k * 128 + (n - 128)]);
  }
}

// ---------------------------------------------------------------------------
// Kernel 2: per-node rotation frames R[a][b] (columns e1,e2,e3)
// ---------------------------------------------------------------------------
__global__ void frames_kernel(const float* pos, char* ws) {
  int i = blockIdx.x * blockDim.x + threadIdx.x;
  if (i >= N_NODES) return;
  float nn[3], ca[3], cc[3];
#pragma unroll
  for (int d = 0; d < 3; ++d) {
    nn[d] = pos[i * 15 + d];
    ca[d] = pos[i * 15 + 3 + d];
    cc[d] = pos[i * 15 + 6 + d];
  }
  float e1[3], e2[3], e3[3];
  float s = 0.f;
#pragma unroll
  for (int d = 0; d < 3; ++d) { e1[d] = cc[d] - ca[d]; s += e1[d] * e1[d]; }
  float inv = 1.f / (sqrtf(s) + 1e-8f);
#pragma unroll
  for (int d = 0; d < 3; ++d) e1[d] *= inv;
  float v2[3], dot = 0.f;
#pragma unroll
  for (int d = 0; d < 3; ++d) { v2[d] = nn[d] - ca[d]; dot += v2[d] * e1[d]; }
  s = 0.f;
#pragma unroll
  for (int d = 0; d < 3; ++d) { e2[d] = v2[d] - dot * e1[d]; s += e2[d] * e2[d]; }
  inv = 1.f / (sqrtf(s) + 1e-8f);
#pragma unroll
  for (int d = 0; d < 3; ++d) e2[d] *= inv;
  e3[0] = e1[1] * e2[2] - e1[2] * e2[1];
  e3[1] = e1[2] * e2[0] - e1[0] * e2[2];
  e3[2] = e1[0] * e2[1] - e1[1] * e2[0];
  float* R = (float*)(ws + WS_R) + (size_t)i * 9;
#pragma unroll
  for (int a = 0; a < 3; ++a) {
    R[a * 3 + 0] = e1[a];
    R[a * 3 + 1] = e2[a];
    R[a * 3 + 2] = e3[a];
  }
}

// ---------------------------------------------------------------------------
// Kernel 3: locIJ = local(8192x256) @ [Wloci|Wlocj](256x256) via WMMA
//   one 16x16 tile per wave; 8 waves/block; 1024 blocks
// ---------------------------------------------------------------------------
__global__ __launch_bounds__(256) void locproj_kernel(const float* local, char* ws) {
  const _Float16* wloc = (const _Float16*)(ws + WS_WLOC);
  float* locij = (float*)(ws + WS_LOCIJ);
  int lane = threadIdx.x & 31;
  int wave = threadIdx.x >> 5;
  int gid = blockIdx.x * 8 + wave;       // 0..8191 tiles
  int mt = gid >> 4, nt = gid & 15;

  v8f acc = zero8();
#pragma unroll
  for (int ks = 0; ks < 8; ++ks) {
    // A fragment: convert f32 -> f16 in registers (2 contiguous 32B runs)
    v16h a;
    const float* ap = local + (mt * 16 + (lane & 15)) * DLOC
                            + ks * 32 + ((lane >> 4) << 3);
#pragma unroll
    for (int j = 0; j < 8; ++j) a[j] = (_Float16)ap[j];
#pragma unroll
    for (int j = 8; j < 16; ++j) a[j] = (_Float16)ap[j + 8];
    v16h b = frag_b_swz(wloc, nt * 8 + ks, lane);   // K/32 = 8
    acc = wmma16(a, b, acc);
  }
  int moff = (lane >= 16) ? 8 : 0;
  int n = nt * 16 + (lane & 15);
#pragma unroll
  for (int r = 0; r < 8; ++r)
    locij[(size_t)(mt * 16 + r + moff) * 256 + n] = acc[r];
}

// ---------------------------------------------------------------------------
// Kernel 4: fused pair features -> LayerNorm -> MLP -> log_softmax
//   one block (8 waves) per node i; all GEMMs via WMMA from LDS
// ---------------------------------------------------------------------------
__global__ __launch_bounds__(256)
void fused_pair_kernel(const float* pos, const int* neighbours, const int* resi,
                       const int* chain, const int* batch, const unsigned char* mask,
                       const float* Wrelpos, const float* ln_scale, const float* ln_bias,
                       const float* b1, const float* b2, const char* ws, float* out) {
  extern __shared__ char smem[];
  _Float16* sWfeat = (_Float16*)(smem + SM_WFEAT);
  _Float16* sW1    = (_Float16*)(smem + SM_W1);
  _Float16* sW2    = (_Float16*)(smem + SM_W2);
  _Float16* sFeatA = (_Float16*)(smem + SM_FEATA);
  float*    sPairF = (float*)   (smem + SM_PAIRF);
  _Float16* sPairH = (_Float16*)(smem + SM_PAIRH);
  _Float16* sHid   = (_Float16*)(smem + SM_HID);
  float*    sLogit = (float*)   (smem + SM_LOGIT);
  int*      sJ     = (int*)     (smem + SM_J);
  int*      sCode  = (int*)     (smem + SM_CODE);
  float*    sPM    = (float*)   (smem + SM_PM);

  const float* R     = (const float*)(ws + WS_R);
  const float* locij = (const float*)(ws + WS_LOCIJ);
  const int i = blockIdx.x;
  const int tid = threadIdx.x, lane = tid & 31, wave = tid >> 5;

  // ---- stage f16 weights (contiguous 112 KB of ws) into LDS ----
  {
    const uint4* src = (const uint4*)ws;
    uint4* dst = (uint4*)smem;
    for (int t = tid; t < STAGE_VEC16; t += 256) dst[t] = src[t];
  }

  // ---- geometry: one lane per neighbour builds 32x64 f16 feature tile ----
  if (tid < KNBR) {
    const int k = tid;
    int nb = neighbours[i * KNBR + k];
    int j = nb < 0 ? 0 : nb;
    sJ[k] = j;
    sPM[k] = (mask[i] ? 1.f : 0.f) * (mask[j] ? 1.f : 0.f) * (nb != -1 ? 1.f : 0.f);
    int rel = resi[j] - resi[i];
    rel = rel < -RELCLIP ? -RELCLIP : (rel > RELCLIP ? RELCLIP : rel);
    bool same = (chain[i] == chain[j]) && (batch[i] == batch[j]);
    sCode[k] = same ? (rel + RELCLIP) : (RELDIM - 1);

    float Ri[9], Rj[9], cai[3], dv[3];
#pragma unroll
    for (int t = 0; t < 9; ++t) { Ri[t] = R[(size_t)i * 9 + t]; Rj[t] = R[(size_t)j * 9 + t]; }
    float dd = 0.f;
#pragma unroll
    for (int d = 0; d < 3; ++d) {
      cai[d] = pos[i * 15 + 3 + d];
      dv[d] = pos[j * 15 + 3 + d] - cai[d];
      dd += dv[d] * dv[d];
    }
    float dist = sqrtf(dd);
    float inv = 1.f / (dist + 1e-8f);
    float u[3] = {dv[0] * inv, dv[1] * inv, dv[2] * inv};

    _Float16* fr = sFeatA + k * 64;
    const float step = 22.f / 15.f;            // linspace(0,22,16)
#pragma unroll
    for (int r = 0; r < 16; ++r) {             // RBF, sigma = 22/16
      float c = dist - step * (float)r;
      fr[r] = (_Float16)__expf(-c * c / 3.78125f);
    }
#pragma unroll
    for (int bb = 0; bb < 3; ++bb)             // f_dir = R_i^T u
      fr[16 + bb] = (_Float16)(Ri[bb] * u[0] + Ri[3 + bb] * u[1] + Ri[6 + bb] * u[2]);
#pragma unroll
    for (int bb = 0; bb < 3; ++bb)             // f_rot = R_i^T R_j
#pragma unroll
      for (int cc = 0; cc < 3; ++cc)
        fr[19 + bb * 3 + cc] = (_Float16)(Ri[bb] * Rj[cc] + Ri[3 + bb] * Rj[3 + cc]
                                        + Ri[6 + bb] * Rj[6 + cc]);
#pragma unroll
    for (int m = 0; m < 5; ++m) {              // f_pv = R_i^T (pos_j[m]-ca_i)
      float p0 = pos[j * 15 + m * 3 + 0] - cai[0];
      float p1 = pos[j * 15 + m * 3 + 1] - cai[1];
      float p2 = pos[j * 15 + m * 3 + 2] - cai[2];
#pragma unroll
      for (int bb = 0; bb < 3; ++bb)
        fr[28 + m * 3 + bb] = (_Float16)(Ri[bb] * p0 + Ri[3 + bb] * p1 + Ri[6 + bb] * p2);
    }
#pragma unroll
    for (int z = 43; z < 64; ++z) fr[z] = (_Float16)0.f;
  }
  __syncthreads();

  // ---- GEMM-0: pair(32x128) = featA(32x64) @ Wfeat(64x128) ----
#pragma unroll
  for (int q = 0; q < 2; ++q) {
    int t = wave * 2 + q;
    int mt = t >> 3, nt = t & 7;
    v8f acc = zero8();
#pragma unroll
    for (int ks = 0; ks < 2; ++ks) {
      v16h a = frag_a(sFeatA, 64, mt * 16, ks * 32, lane);
      v16h b = frag_b_swz(sWfeat, nt * 2 + ks, lane);   // K/32 = 2
      acc = wmma16(a, b, acc);
    }
    int moff = (lane >= 16) ? 8 : 0;
    int n = nt * 16 + (lane & 15);
#pragma unroll
    for (int r = 0; r < 8; ++r)
      sPairF[(mt * 16 + r + moff) * PP + n] = acc[r];
  }
  __syncthreads();

  // ---- add locI[i] + locJ[j] + W_relpos[code], apply pair mask ----
  for (int e = tid; e < KNBR * PP; e += 256) {
    int m = e >> 7, c = e & 127;
    int j = sJ[m];
    float v = sPairF[e]
            + locij[(size_t)i * 256 + c]
            + locij[(size_t)j * 256 + 128 + c]
            + Wrelpos[sCode[m] * PP + c];
    sPairF[e] = v * sPM[m];
  }
  __syncthreads();

  // ---- LayerNorm over P=128 (4 rows per wave, wave32 shfl reductions) ----
#pragma unroll
  for (int rr = 0; rr < 4; ++rr) {
    int m = wave * 4 + rr;
    float x[4], s = 0.f, sq = 0.f;
#pragma unroll
    for (int q = 0; q < 4; ++q) {
      x[q] = sPairF[m * PP + lane * 4 + q];
      s += x[q]; sq += x[q] * x[q];
    }
#pragma unroll
    for (int o = 16; o > 0; o >>= 1) {
      s  += __shfl_xor(s, o, 32);
      sq += __shfl_xor(sq, o, 32);
    }
    float mu = s * (1.f / PP);
    float var = sq * (1.f / PP) - mu * mu;
    float rs = rsqrtf(var + 1e-5f);
#pragma unroll
    for (int q = 0; q < 4; ++q) {
      int c = lane * 4 + q;
      sPairH[m * PP + c] = (_Float16)((x[q] - mu) * rs * ln_scale[c] + ln_bias[c]);
    }
  }
  __syncthreads();

  // ---- GEMM-1: hidden(32x256) = gelu(pair @ W1 + b1) ----
#pragma unroll
  for (int q = 0; q < 4; ++q) {
    int t = wave * 4 + q;
    int mt = t >> 4, nt = t & 15;
    v8f acc = zero8();
#pragma unroll
    for (int ks = 0; ks < 4; ++ks) {
      v16h a = frag_a(sPairH, PP, mt * 16, ks * 32, lane);
      v16h b = frag_b_swz(sW1, nt * 4 + ks, lane);      // K/32 = 4
      acc = wmma16(a, b, acc);
    }
    int n = nt * 16 + (lane & 15);
    float bv = b1[n];
    int moff = (lane >= 16) ? 8 : 0;
#pragma unroll
    for (int r = 0; r < 8; ++r) {
      float xg = acc[r] + bv;
      float g = 0.5f * xg * (1.f + tanhf(0.7978845608f * (xg + 0.044715f * xg * xg * xg)));
      sHid[(mt * 16 + r + moff) * 256 + n] = (_Float16)g;
    }
  }
  __syncthreads();

  // ---- GEMM-2: logits(32x64) = hidden @ W2 + b2 (one tile per wave) ----
  {
    int mt = wave >> 2, nt = wave & 3;
    v8f acc = zero8();
#pragma unroll
    for (int ks = 0; ks < 8; ++ks) {
      v16h a = frag_a(sHid, 256, mt * 16, ks * 32, lane);
      v16h b = frag_b_swz(sW2, nt * 8 + ks, lane);      // K/32 = 8
      acc = wmma16(a, b, acc);
    }
    int n = nt * 16 + (lane & 15);
    float bv = b2[n];
    int moff = (lane >= 16) ? 8 : 0;
#pragma unroll
    for (int r = 0; r < 8; ++r)
      sLogit[(mt * 16 + r + moff) * BINS + n] = acc[r] + bv;
  }
  __syncthreads();

  // ---- log_softmax over 64 bins (4 rows per wave, 2 bins per lane) ----
#pragma unroll
  for (int rr = 0; rr < 4; ++rr) {
    int m = wave * 4 + rr;
    float x0 = sLogit[m * BINS + lane];
    float x1 = sLogit[m * BINS + lane + 32];
    float mx = fmaxf(x0, x1);
#pragma unroll
    for (int o = 16; o > 0; o >>= 1) mx = fmaxf(mx, __shfl_xor(mx, o, 32));
    float se = __expf(x0 - mx) + __expf(x1 - mx);
#pragma unroll
    for (int o = 16; o > 0; o >>= 1) se += __shfl_xor(se, o, 32);
    float lse = mx + logf(se);
    float* op = out + ((size_t)i * KNBR + m) * BINS;
    op[lane] = x0 - lse;
    op[lane + 32] = x1 - lse;
  }
}

// ---------------------------------------------------------------------------
extern "C" void kernel_launch(void* const* d_in, const int* in_sizes, int n_in,
                              void* d_out, int out_size, void* d_ws, size_t ws_size,
                              hipStream_t stream) {
  const float* local      = (const float*)d_in[0];
  const float* pos        = (const float*)d_in[1];
  const int*   neighbours = (const int*)d_in[2];
  const int*   resi       = (const int*)d_in[3];
  const int*   chain      = (const int*)d_in[4];
  const int*   batch      = (const int*)d_in[5];
  const unsigned char* mask = (const unsigned char*)d_in[6];
  const float* Wrelpos    = (const float*)d_in[7];
  const float* Wdist      = (const float*)d_in[8];
  const float* Wdir       = (const float*)d_in[9];
  const float* Wrot       = (const float*)d_in[10];
  const float* Wpvec      = (const float*)d_in[11];
  const float* Wloci      = (const float*)d_in[12];
  const float* Wlocj      = (const float*)d_in[13];
  const float* ln_scale   = (const float*)d_in[14];
  const float* ln_bias    = (const float*)d_in[15];
  const float* W1         = (const float*)d_in[16];
  const float* b1         = (const float*)d_in[17];
  const float* W2         = (const float*)d_in[18];
  const float* b2         = (const float*)d_in[19];
  char* ws = (char*)d_ws;
  float* out = (float*)d_out;

  // idempotent + deterministic; required for >64KB dynamic LDS on some stacks
  (void)hipFuncSetAttribute((const void*)fused_pair_kernel,
                            hipFuncAttributeMaxDynamicSharedMemorySize, SM_BYTES);

  prep_weights<<<64, 256, 0, stream>>>(Wdist, Wdir, Wrot, Wpvec, Wloci, Wlocj,
                                       W1, W2, ws);
  frames_kernel<<<N_NODES / 256, 256, 0, stream>>>(pos, ws);
  locproj_kernel<<<1024, 256, 0, stream>>>(local, ws);
  fused_pair_kernel<<<N_NODES, 256, SM_BYTES, stream>>>(
      pos, neighbours, resi, chain, batch, mask, Wrelpos, ln_scale, ln_bias,
      b1, b2, ws, out);
}
